// GCN_ChPt_50113678409892
// MI455X (gfx1250) — matrix operand order repaired and checked by reading, not compiled, not run on hardware
//
#include <hip/hip_runtime.h>

// ---------------------------------------------------------------------------
// GCN (32->64->64->1) + linear + sigmoid*pmax for MI455X (gfx1250, wave32).
// Dense matmuls use V_WMMA_F32_16X16X4_F32 (fp32-exact); falls back to
// V_WMMA_F32_16X16X32_F16 if the fp32 builtin is not declared.
// Edge aggregation is the bandwidth-bound part; all hot buffers fit in the
// 192MB L2, norm[e] is precomputed once and reused for all three layers.
// ---------------------------------------------------------------------------

typedef __attribute__((ext_vector_type(2)))  float    v2f;
typedef __attribute__((ext_vector_type(8)))  float    v8f;
typedef __attribute__((ext_vector_type(16))) _Float16 v16h;

#if defined(__AMDGCN__) && __has_builtin(__builtin_amdgcn_wmma_f32_16x16x4_f32)
#define HAVE_WMMA_F32 1
#else
#define HAVE_WMMA_F32 0
#endif

#define TPB 256

// ---------------- degree / normalization ----------------

__global__ __launch_bounds__(TPB) void k_init_one(float* deg, int n) {
  int i = blockIdx.x * TPB + threadIdx.x;
  if (i < n) deg[i] = 1.0f;                       // self-loop weight
}

__global__ __launch_bounds__(TPB) void k_deg_accum(const float* __restrict__ w,
                                                   const int* __restrict__ dst,
                                                   float* deg, int e) {
  int i = blockIdx.x * TPB + threadIdx.x;
  if (i < e) atomicAdd(&deg[dst[i]], w[i]);
}

__global__ __launch_bounds__(TPB) void k_rsqrt(float* deg, int n) {
  int i = blockIdx.x * TPB + threadIdx.x;
  if (i < n) deg[i] = rsqrtf(deg[i]);             // deg >= 1 always
}

__global__ __launch_bounds__(TPB) void k_norm(const float* __restrict__ w,
                                              const int* __restrict__ src,
                                              const int* __restrict__ dst,
                                              const float* __restrict__ dinv,
                                              float* nrm, int e) {
  int i = blockIdx.x * TPB + threadIdx.x;
  if (i < e) nrm[i] = dinv[src[i]] * w[i] * dinv[dst[i]];
}

// ---------------- dense GEMM via WMMA ----------------
// C[N,64] = A[N,KDIM] * W[KDIM,64]; 8 waves/block, wave owns a 16x16 tile.
// Requires nrows % 32 == 0 (N = 100000 = 32 * 3125).

template <int KDIM>
__global__ __launch_bounds__(TPB) void k_gemm_wmma(const float* __restrict__ A,
                                                   const float* __restrict__ W,
                                                   float* __restrict__ C, int nrows) {
  const int wave = threadIdx.x >> 5;
  const int lane = threadIdx.x & 31;
  const int row0 = blockIdx.x * 32 + (wave >> 2) * 16;
  const int col0 = (wave & 3) * 16;
  if (row0 + 16 > nrows) return;                  // wave-uniform; EXEC stays full
  const int lr  = lane & 15;
  const bool hi = lane >= 16;
  v8f acc = {};
#if HAVE_WMMA_F32
  // fp32 WMMA: A 16x4 frag = 2 VGPR (lanes0-15: K=k,k+1; lanes16-31: K=k+2,k+3)
  #pragma unroll
  for (int k = 0; k < KDIM; k += 4) {
    const int ka = k + (hi ? 2 : 0);
    const float* arow = A + (size_t)(row0 + lr) * KDIM + ka;
    v2f a, b;
    a.x = arow[0];
    a.y = arow[1];
    b.x = W[(size_t)(ka + 0) * 64 + col0 + lr];
    b.y = W[(size_t)(ka + 1) * 64 + col0 + lr];
    acc = __builtin_amdgcn_wmma_f32_16x16x4_f32(false, a, false, b, (short)0,
                                                acc, false, false);
  }
#else
  // f16 fallback: A 16x32 frag per ISA layout, B 32x16 in the same K order as A
  #pragma unroll
  for (int k = 0; k < KDIM; k += 32) {
    const float* arow = A + (size_t)(row0 + lr) * KDIM + k;
    v16h a, b;
    #pragma unroll
    for (int h = 0; h < 16; ++h) {
      int kk = (h < 8) ? h : 8 + h;
      if (hi) kk += 8;
      a[h] = (_Float16)arow[kk];
      b[h] = (_Float16)W[(size_t)(k + kk) * 64 + col0 + lr];
    }
    acc = __builtin_amdgcn_wmma_f32_16x16x32_f16(false, a, false, b, (short)0,
                                                 acc, false, false);
  }
#endif
  const int crow = row0 + (hi ? 8 : 0);           // C VGPR r: M = r (+8 for hi half)
  #pragma unroll
  for (int r = 0; r < 8; ++r)
    C[(size_t)(crow + r) * 64 + col0 + lr] = acc[r];
}

// ---------------- aggregation (64-wide) ----------------

__global__ __launch_bounds__(TPB) void k_self_init64(const float* __restrict__ hlin,
                                                     const float* __restrict__ dinv,
                                                     float* agg, int n) {
  int t = blockIdx.x * TPB + threadIdx.x;
  if (t >= n * 64) return;
  float di = dinv[t >> 6];
  agg[t] = di * di * hlin[t];                     // self-loop contribution
}

__global__ __launch_bounds__(TPB) void k_scatter64(const float* __restrict__ hlin,
                                                   const float* __restrict__ nrm,
                                                   const int* __restrict__ src,
                                                   const int* __restrict__ dst,
                                                   float* agg, int e) {
  int t = blockIdx.x * TPB + threadIdx.x;
  int ed = t >> 4;
  if (ed >= e) return;
  int q = t & 15;
  int s = src[ed], d = dst[ed];
  float nm = nrm[ed];
  float4 v = ((const float4*)hlin)[(size_t)s * 16 + q];
  float* o = agg + (size_t)d * 64 + q * 4;
  atomicAdd(o + 0, nm * v.x);
  atomicAdd(o + 1, nm * v.y);
  atomicAdd(o + 2, nm * v.z);
  atomicAdd(o + 3, nm * v.w);
}

__global__ __launch_bounds__(TPB) void k_bias_relu64(const float* __restrict__ agg,
                                                     const float* __restrict__ b,
                                                     float* h, int n, int relu) {
  int t = blockIdx.x * TPB + threadIdx.x;
  if (t >= n * 64) return;
  float v = agg[t] + b[t & 63];
  h[t] = relu ? fmaxf(v, 0.0f) : v;
}

// ---------------- final scalar layer ----------------

__global__ __launch_bounds__(TPB) void k_dot_w2(const float* __restrict__ h2,
                                                const float* __restrict__ w2,
                                                float* z, int n) {
  int i = blockIdx.x * TPB + threadIdx.x;
  if (i >= n) return;
  const float4* hv = (const float4*)(h2 + (size_t)i * 64);
  const float4* wv = (const float4*)w2;           // W2 is [64,1] contiguous
  float acc = 0.0f;
  #pragma unroll
  for (int q = 0; q < 16; ++q) {
    float4 a = hv[q], b = wv[q];
    acc += a.x * b.x + a.y * b.y + a.z * b.z + a.w * b.w;
  }
  z[i] = acc;
}

__global__ __launch_bounds__(TPB) void k_self_init1(const float* __restrict__ z,
                                                    const float* __restrict__ dinv,
                                                    float* zagg, int n) {
  int i = blockIdx.x * TPB + threadIdx.x;
  if (i >= n) return;
  float di = dinv[i];
  zagg[i] = di * di * z[i];
}

__global__ __launch_bounds__(TPB) void k_scatter1(const float* __restrict__ z,
                                                  const float* __restrict__ nrm,
                                                  const int* __restrict__ src,
                                                  const int* __restrict__ dst,
                                                  float* zagg, int e) {
  int i = blockIdx.x * TPB + threadIdx.x;
  if (i < e) atomicAdd(&zagg[dst[i]], nrm[i] * z[src[i]]);
}

__global__ __launch_bounds__(TPB) void k_final(const float* __restrict__ zagg,
                                               const float* __restrict__ pmax,
                                               const float* __restrict__ b2,
                                               const float* __restrict__ lw,
                                               const float* __restrict__ lb,
                                               float* out, int n) {
  int i = blockIdx.x * TPB + threadIdx.x;
  if (i >= n) return;
  float v = (zagg[i] + b2[0]) * lw[0] + lb[0];
  out[i] = pmax[i] / (1.0f + expf(-v));
}

// ---------------- host-side orchestration ----------------

extern "C" void kernel_launch(void* const* d_in, const int* in_sizes, int n_in,
                              void* d_out, int out_size, void* d_ws, size_t ws_size,
                              hipStream_t stream) {
  const float* x0   = (const float*)d_in[0];
  const float* pmax = (const float*)d_in[1];
  const int*   ei   = (const int*)d_in[2];   // edge_index [2,E] (int indices)
  const float* ew   = (const float*)d_in[3];
  const float* W0   = (const float*)d_in[4];
  const float* b0   = (const float*)d_in[5];
  const float* W1   = (const float*)d_in[6];
  const float* b1   = (const float*)d_in[7];
  const float* W2   = (const float*)d_in[8];
  const float* b2   = (const float*)d_in[9];
  const float* lw   = (const float*)d_in[10];
  const float* lb   = (const float*)d_in[11];
  float* out = (float*)d_out;

  const int N = in_sizes[1];                 // pmax: N elements
  const int E = in_sizes[3];                 // edge_weights: E elements
  const int* src = ei;
  const int* dst = ei + E;

  // Workspace layout (floats): dinv[N] | norm[E] | bufA[N*64] | bufB[N*64]
  float* ws   = (float*)d_ws;
  float* dinv = ws;
  float* nrm  = dinv + N;
  float* bufA = nrm + E;
  float* bufB = bufA + (size_t)N * 64;
  float* z    = bufA;                        // reuse, disjoint from zagg
  float* zagg = bufA + N;

  const int gN   = (N + TPB - 1) / TPB;
  const int gE   = (E + TPB - 1) / TPB;
  const int gN64 = (N * 64 + TPB - 1) / TPB;
  const int gE16 = (E * 16 + TPB - 1) / TPB;
  const int gMM  = (N + 31) / 32;            // 8 waves/block -> 32 rows/block

  // degree -> dinv -> per-edge norm (shared by all three layers)
  k_init_one <<<gN, TPB, 0, stream>>>(dinv, N);
  k_deg_accum<<<gE, TPB, 0, stream>>>(ew, dst, dinv, E);
  k_rsqrt    <<<gN, TPB, 0, stream>>>(dinv, N);
  k_norm     <<<gE, TPB, 0, stream>>>(ew, src, dst, dinv, nrm, E);

  // layer 0: x0 @ W0 -> bufA ; aggregate -> bufB ; relu(+b0) -> bufA
  k_gemm_wmma<32><<<gMM, TPB, 0, stream>>>(x0, W0, bufA, N);
  k_self_init64 <<<gN64, TPB, 0, stream>>>(bufA, dinv, bufB, N);
  k_scatter64   <<<gE16, TPB, 0, stream>>>(bufA, nrm, src, dst, bufB, E);
  k_bias_relu64 <<<gN64, TPB, 0, stream>>>(bufB, b0, bufA, N, 1);

  // layer 1: h1 @ W1 -> bufB ; aggregate -> bufA ; relu(+b1) -> bufB
  k_gemm_wmma<64><<<gMM, TPB, 0, stream>>>(bufA, W1, bufB, N);
  k_self_init64 <<<gN64, TPB, 0, stream>>>(bufB, dinv, bufA, N);
  k_scatter64   <<<gE16, TPB, 0, stream>>>(bufB, nrm, src, dst, bufA, E);
  k_bias_relu64 <<<gN64, TPB, 0, stream>>>(bufA, b1, bufB, N, 1);

  // layer 2: h2 . W2 -> z ; scalar aggregate ; final linear+sigmoid*pmax
  k_dot_w2     <<<gN, TPB, 0, stream>>>(bufB, W2, z, N);
  k_self_init1 <<<gN, TPB, 0, stream>>>(z, dinv, zagg, N);
  k_scatter1   <<<gE, TPB, 0, stream>>>(z, nrm, src, dst, zagg, E);
  k_final      <<<gN, TPB, 0, stream>>>(zagg, pmax, b2, lw, lb, out, N);
}
